// LatencyPredictor_75170517615303
// MI455X (gfx1250) — matrix operand order repaired and checked by reading, not compiled, not run on hardware
//
#include <hip/hip_runtime.h>

typedef __attribute__((ext_vector_type(16))) _Float16 v16h;
typedef __attribute__((ext_vector_type(8)))  _Float16 v8h;
typedef __attribute__((ext_vector_type(8)))  float    v8f;

#define B_   64
#define L_   8192
#define T_   256
#define K1_  64
#define K2_  32
#define LW_  (L_ - T_ + 1)          // 7937 windows
#define LP_  (L_ + T_)              // 8448 padded row length (f16)
#define MT_  ((LW_ + 15) / 16)      // 497 M-tiles of 16 rows
#define MPW  2                      // M-tiles per wave (32 rows, Hankel A reuse)
#define PAIRS ((MT_ + MPW - 1) / MPW)   // 249
#define WPB  8                      // waves per block (256 threads, wave32)
#define NBX  ((PAIRS + WPB - 1) / WPB)  // 32

// workspace layout (halves, then floats)
#define XH_HALVES  (2 * B_ * LP_)            // padded f16 x, both chains
#define W1P_HALVES (2 * 8 * 4 * 32 * 16)     // W1 pre-swizzled to WMMA B layout
#define W2P_HALVES (2 * 2 * 2 * 32 * 16)     // W2 pre-swizzled to WMMA B layout
#define VALT_OFF_B 2236416                   // byte offset of valT (256-aligned)

// output layout per chain (f32 elements): da | ida | vala | la
#define DA_N      ((size_t)B_ * LW_ * K2_)
#define IDA_N     ((size_t)B_ * LW_)
#define CHAIN_OUT (DA_N + 2 * IDA_N + (size_t)B_)

// ---------------------------------------------------------------------------
// Prep 1: x (f32) -> padded f16 rows (zero tail so Hankel tiles never guard)
// ---------------------------------------------------------------------------
__global__ void pack_x_kernel(const float* __restrict__ xa,
                              const float* __restrict__ xb,
                              _Float16* __restrict__ xh) {
    int i = blockIdx.x * blockDim.x + threadIdx.x;
    if (i >= XH_HALVES) return;
    int chain = i / (B_ * LP_);
    int r     = i % (B_ * LP_);
    int b     = r / LP_;
    int j     = r % LP_;
    const float* x = chain ? xb : xa;
    xh[i] = (j < L_) ? (_Float16)x[(size_t)b * L_ + j] : (_Float16)0.0f;
}

// ---------------------------------------------------------------------------
// Prep 2: swizzle W1 [256x64] and W2 [64x32] into wave32 WMMA B-operand
// layout: per (kchunk, ntile, lane) a lane-contiguous run of 16 halves;
// half h of lane group g holds K = 8g + h (h<8), 16 + 8g + (h-8) (h>=8) (n = lane&15).
// ---------------------------------------------------------------------------
__global__ void pack_w_kernel(const float* __restrict__ W1a, const float* __restrict__ W1b,
                              const float* __restrict__ W2a, const float* __restrict__ W2b,
                              _Float16* __restrict__ w1p, _Float16* __restrict__ w2p) {
    int i = blockIdx.x * blockDim.x + threadIdx.x;
    if (i < W1P_HALVES) {
        int chain = i / (W1P_HALVES / 2);
        int r     = i % (W1P_HALVES / 2);
        int slot = r >> 4, h = r & 15;
        int lane = slot & 31;
        int ct   = slot >> 5;
        int t = ct & 3, c = ct >> 2;           // 4 N-tiles, 8 K-chunks
        int g = lane >> 4;
        int k = c * 32 + (h & 7) + 8 * g + 16 * (h >> 3);
        int n = t * 16 + (lane & 15);
        const float* w = chain ? W1b : W1a;
        w1p[i] = (_Float16)w[k * K1_ + n];
    } else if (i < W1P_HALVES + W2P_HALVES) {
        int i2    = i - W1P_HALVES;
        int chain = i2 / (W2P_HALVES / 2);
        int r     = i2 % (W2P_HALVES / 2);
        int slot = r >> 4, h = r & 15;
        int lane = slot & 31;
        int ct   = slot >> 5;
        int t = ct & 1, c = ct >> 1;           // 2 N-tiles, 2 K-chunks
        int g = lane >> 4;
        int k = c * 32 + (h & 7) + 8 * g + 16 * (h >> 3);
        int n = t * 16 + (lane & 15);
        const float* w = chain ? W2b : W2a;
        w2p[i2] = (_Float16)w[k * K2_ + n];
    }
}

// ---------------------------------------------------------------------------
// Main: implicit-GEMM dendrite MLP. One wave = 32 window rows (2 M-tiles
// sharing every B operand; Hankel overlap shares A registers) x full N.
// GEMM1 (K=256, f16 WMMA, f32 acc) -> relu -> LDS f16 -> GEMM2 (K=64)
// -> relu -> store da, shfl max/argmax -> vala/ida (+ transposed valT).
// ---------------------------------------------------------------------------
__global__ __launch_bounds__(256) void dendrite_gemm_kernel(
    const _Float16* __restrict__ xh,
    const _Float16* __restrict__ w1p,
    const _Float16* __restrict__ w2p,
    const float* __restrict__ b1a, const float* __restrict__ b1b,
    const float* __restrict__ b2a, const float* __restrict__ b2b,
    float* __restrict__ out, float* __restrict__ valT)
{
    __shared__ __align__(32) _Float16 w1s[8 * 4 * 32 * 16];        // 32 KB
    __shared__ __align__(32) _Float16 w2s[2 * 2 * 32 * 16];        //  4 KB
    __shared__ __align__(32) _Float16 d1s[WPB][MPW * 16 * 64];     // 32 KB

    const int tid   = threadIdx.x;
    const int chain = blockIdx.z;
    const int b     = blockIdx.y;

    // stage pre-swizzled weights into LDS (all threads, before any early-out)
    {
        const uint32_t* s1 = (const uint32_t*)(w1p + (size_t)chain * (W1P_HALVES / 2));
        uint32_t*       q1 = (uint32_t*)w1s;
        for (int i = tid; i < (W1P_HALVES / 2) / 2; i += 256) q1[i] = s1[i];
        const uint32_t* s2 = (const uint32_t*)(w2p + (size_t)chain * (W2P_HALVES / 2));
        uint32_t*       q2 = (uint32_t*)w2s;
        for (int i = tid; i < (W2P_HALVES / 2) / 2; i += 256) q2[i] = s2[i];
    }
    __syncthreads();

    const int wave = tid >> 5;
    const int lane = tid & 31;
    const int pair = blockIdx.x * WPB + wave;
    if (pair >= PAIRS) return;             // whole wave exits: EXEC all-ones for WMMA

    const int m  = lane & 15;              // row (A) / column n (B,C,D) within group
    const int g  = lane >> 4;              // lane group
    const int l0 = pair * (MPW * 16);

    const float* b1 = chain ? b1b : b1a;
    const float* b2 = chain ? b2b : b2a;
    float b1v[4], b2v[2];
    #pragma unroll
    for (int t = 0; t < 4; ++t) b1v[t] = b1[t * 16 + m];
    #pragma unroll
    for (int t = 0; t < 2; ++t) b2v[t] = b2[t * 16 + m];

    // ---- GEMM1: two [16 x 256] Hankel tiles  @  W1 [256 x 64] ----
    // A layout (16-bit 16x32): lane(m,g) halves 0..7 -> K = kc*32+g*8+0..7,
    // halves 8..15 -> +16.  Hankel overlap: tile1.h[0] == tile0.h[1], and
    // chunk kc+1's first v8h == chunk kc's last -> sliding 3-register window.
    v8f c1[MPW][4] = {};
    const _Float16* xrow = xh + (size_t)(chain * B_ + b) * LP_ + l0 + m;
    v8h P0 = *(const v8h*)(xrow + g * 8);
    #pragma unroll
    for (int kc = 0; kc < 8; ++kc) {
        const _Float16* ap = xrow + kc * 32 + g * 8;
        v8h P1 = *(const v8h*)(ap + 16);
        v8h P2 = *(const v8h*)(ap + 32);
        union { v16h v; v8h h[2]; } a0, a1;
        a0.h[0] = P0; a0.h[1] = P1;        // rows l0..l0+15
        a1.h[0] = P1; a1.h[1] = P2;        // rows l0+16..l0+31
        #pragma unroll
        for (int t = 0; t < 4; ++t) {
            const v16h bm = *(const v16h*)(w1s + ((kc * 4 + t) * 32 + lane) * 16);
            c1[0][t] = __builtin_amdgcn_wmma_f32_16x16x32_f16(
                           false, a0.v, false, bm, (short)0, c1[0][t], false, false);
            c1[1][t] = __builtin_amdgcn_wmma_f32_16x16x32_f16(
                           false, a1.v, false, bm, (short)0, c1[1][t], false, false);
        }
        P0 = P2;
    }

    // bias + relu, stage d1 tiles as f16 row-major [32][64] in this wave's slab
    _Float16* dtile = d1s[wave];
    #pragma unroll
    for (int s = 0; s < MPW; ++s)
        #pragma unroll
        for (int t = 0; t < 4; ++t)
            #pragma unroll
            for (int r = 0; r < 8; ++r) {
                float v = c1[s][t][r] + b1v[t];
                v = v > 0.0f ? v : 0.0f;
                dtile[s * 1024 + (r + 8 * g) * 64 + t * 16 + m] = (_Float16)v;
            }
    // same-wave LDS ops are in order; loads below get compiler s_wait_dscnt

    // ---- GEMM2: d1 [32 x 64] @ W2 [64 x 32], C pre-seeded with b2 ----
    v8f c2[MPW][2];
    #pragma unroll
    for (int s = 0; s < MPW; ++s)
        #pragma unroll
        for (int t = 0; t < 2; ++t)
            #pragma unroll
            for (int r = 0; r < 8; ++r) c2[s][t][r] = b2v[t];

    #pragma unroll
    for (int kc = 0; kc < 2; ++kc) {
        union { v16h v; v8h h[2]; } a0, a1;
        const _Float16* ap0 = dtile + m * 64 + kc * 32 + g * 8;           // 16B aligned
        const _Float16* ap1 = ap0 + 1024;
        a0.h[0] = *(const v8h*)(ap0); a0.h[1] = *(const v8h*)(ap0 + 16);
        a1.h[0] = *(const v8h*)(ap1); a1.h[1] = *(const v8h*)(ap1 + 16);
        #pragma unroll
        for (int t = 0; t < 2; ++t) {
            const v16h bm = *(const v16h*)(w2s + ((kc * 2 + t) * 32 + lane) * 16);
            c2[0][t] = __builtin_amdgcn_wmma_f32_16x16x32_f16(
                           false, a0.v, false, bm, (short)0, c2[0][t], false, false);
            c2[1][t] = __builtin_amdgcn_wmma_f32_16x16x32_f16(
                           false, a1.v, false, bm, (short)0, c2[1][t], false, false);
        }
    }

    // ---- relu, store da, row max/argmax (first-occurrence ties) ----
    float*       daC  = out + (size_t)chain * CHAIN_OUT;
    float*       idaC = daC + DA_N;
    float*       valC = idaC + IDA_N;
    const size_t bRow = (size_t)b * LW_;

    #pragma unroll
    for (int s = 0; s < MPW; ++s) {
        #pragma unroll
        for (int r = 0; r < 8; ++r) {
            const int  row = s * 16 + r + 8 * g;      // C layout: VGPR r, group g -> M
            const int  l   = l0 + row;
            const bool ok  = (l < LW_);
            float v0 = c2[s][0][r]; v0 = v0 > 0.0f ? v0 : 0.0f;   // n = m
            float v1 = c2[s][1][r]; v1 = v1 > 0.0f ? v1 : 0.0f;   // n = 16 + m
            if (ok) {
                float* dp = daC + (bRow + l) * K2_;
                dp[m]      = v0;
                dp[16 + m] = v1;
            }
            float bv = v0; int bi = m;
            if (v1 > bv) { bv = v1; bi = 16 + m; }    // strict > keeps lower n on tie
            #pragma unroll
            for (int off = 1; off < 16; off <<= 1) {  // reduce across 16-lane N group
                float ov = __shfl_xor(bv, off, 32);
                int   oi = __shfl_xor(bi, off, 32);
                if (ov > bv || (ov == bv && oi < bi)) { bv = ov; bi = oi; }
            }
            if (ok && m == r) {
                valC[bRow + l] = bv;
                idaC[bRow + l] = (float)bi;
                valT[(size_t)l * 128 + chain * 64 + b] = bv;  // coalesced LIF staging
            }
        }
    }
}

// ---------------------------------------------------------------------------
// LIF scan: serial in Lw, parallel over 128 (chain, batch) pairs (1 lane each).
// Coalesced reads from valT[l][128]; sigmoid via one v_tanh_f32 per step:
//   sigma(10(v-1)) = 0.5 + 0.5*tanh(5(v-1))
// ---------------------------------------------------------------------------
__global__ void lif_kernel(const float* __restrict__ valT, float* __restrict__ out) {
    int c = blockIdx.x * blockDim.x + threadIdx.x;
    if (c >= 2 * B_) return;
    int chain = c >> 6;
    int b     = c & (B_ - 1);
    float* laC = out + (size_t)chain * CHAIN_OUT + DA_N + 2 * IDA_N;

    float v = 0.0f, best = -1.0f;
    int   bi = 0;
    #pragma unroll 8
    for (int l = 0; l < LW_; ++l) {
        float I = valT[(size_t)l * 128 + c];
        v = __builtin_fmaf(v, 0.95f, I);              // v += -v/TAU + I
        float x = __builtin_fmaf(v, 5.0f, -5.0f);     // 5*(v - 1)
        float th;
        asm volatile("v_tanh_f32 %0, %1" : "=v"(th) : "v"(x));
        float s = __builtin_fmaf(th, 0.5f, 0.5f);     // sigmoid
        v = v * (0.5f - 0.5f * th);                   // v *= (1 - s) soft reset
        if (s > best) { best = s; bi = l; }           // first max
    }
    laC[b] = (float)bi;
}

// ---------------------------------------------------------------------------
extern "C" void kernel_launch(void* const* d_in, const int* in_sizes, int n_in,
                              void* d_out, int out_size, void* d_ws, size_t ws_size,
                              hipStream_t stream) {
    const float* xa  = (const float*)d_in[0];
    const float* xb  = (const float*)d_in[1];
    const float* W1a = (const float*)d_in[2];
    const float* b1a = (const float*)d_in[3];
    const float* W2a = (const float*)d_in[4];
    const float* b2a = (const float*)d_in[5];
    const float* W1b = (const float*)d_in[6];
    const float* b1b = (const float*)d_in[7];
    const float* W2b = (const float*)d_in[8];
    const float* b2b = (const float*)d_in[9];
    float* out = (float*)d_out;

    _Float16* xh   = (_Float16*)d_ws;
    _Float16* w1p  = xh + XH_HALVES;
    _Float16* w2p  = w1p + W1P_HALVES;
    float*    valT = (float*)((char*)d_ws + VALT_OFF_B);   // [LW_][128]

    pack_x_kernel<<<(XH_HALVES + 255) / 256, 256, 0, stream>>>(xa, xb, xh);
    pack_w_kernel<<<(W1P_HALVES + W2P_HALVES + 255) / 256, 256, 0, stream>>>(
        W1a, W1b, W2a, W2b, w1p, w2p);

    dim3 grid(NBX, B_, 2);
    dendrite_gemm_kernel<<<grid, 256, 0, stream>>>(xh, w1p, w2p,
                                                   b1a, b1b, b2a, b2b, out, valT);
    lif_kernel<<<1, 128, 0, stream>>>(valT, out);
}
